// LSTM_84052509982843
// MI455X (gfx1250) — matrix operand order repaired and checked by reading, not compiled, not run on hardware
//
#include <hip/hip_runtime.h>

typedef __attribute__((ext_vector_type(16))) _Float16 v16h;
typedef __attribute__((ext_vector_type(8)))  float    v8f;
typedef __attribute__((ext_vector_type(2)))  __fp16   f16x2;

#define T_STEPS 2048
#define HH 32

__device__ __forceinline__ float fast_exp(float x) {
    return __builtin_amdgcn_exp2f(x * 1.44269504088896341f);
}
__device__ __forceinline__ float fast_rcp(float x) {
    return __builtin_amdgcn_rcpf(x);
}

// Hardware tanh if the gfx1250 builtin exists, else exp2/rcp identity.
__device__ __forceinline__ float tanh_hw(float y) {
#if __has_builtin(__builtin_amdgcn_tanhf)
    return __builtin_amdgcn_tanhf(y);        // single V_TANH_F32
#else
    return 1.0f - 2.0f * fast_rcp(1.0f + fast_exp(2.0f * y));
#endif
}

// Inputs for i/f/o gates are pre-scaled by 0.5 (folded into W_hh/bias/W_ih),
// so sigmoid(orig) = 0.5*tanh(orig/2) + 0.5 = fma(tanh(y), .5, .5).
__device__ __forceinline__ float sigm_scaled(float y) {
    return fmaf(tanh_hw(y), 0.5f, 0.5f);
}

// gate_init = (f16 hi of bw) * x + (f16 lo of bw): one VOP3P v_fma_mix_f32.
// Inline asm keeps bw packed in a single VGPR (compiler otherwise widens to 2x f32).
__device__ __forceinline__ float gate_init(unsigned int bw, float x) {
    float r;
    asm("v_fma_mix_f32 %0, %1, %2, %3 op_sel:[1,0,0] op_sel_hi:[1,0,1]"
        : "=v"(r) : "v"(bw), "v"(x), "v"(bw));
    return r;
}

__device__ __forceinline__ unsigned int pk_f16(float lo, float hi) {
    return __builtin_bit_cast(unsigned int, __builtin_amdgcn_cvt_pkrtz(lo, hi));
}

// One wave processes 16 batch rows for all 2048 timesteps.
// gates^T = W_hh @ h^T via 8x V_WMMA_F32_16X16X32_F16 per step.
__global__ __launch_bounds__(256, 1)
void lstm_wmma_kernel(const float* __restrict__ x,
                      const float* __restrict__ W_ih,
                      const float* __restrict__ W_hh,
                      const float* __restrict__ b_ih,
                      const float* __restrict__ b_hh,
                      const float* __restrict__ W3,
                      const float* __restrict__ b3,
                      const float* __restrict__ W4,
                      const float* __restrict__ b4,
                      float* __restrict__ out)
{
    __shared__ float hbuf[8 * 16 * HH];   // 8 waves * 16 batch * 32 hidden

    const int lane = threadIdx.x & 31;
    const int wave = threadIdx.x >> 5;
    const int half = lane >> 4;           // which 16-lane half
    const int brow = lane & 15;           // A-row (gate) index AND batch column index
    const int b0   = (blockIdx.x * 8 + wave) * 16;

    // ---------------- A tiles: W_hh[128,32] as 8 f16 16x32 A-layout tiles ----
    // Tiles 0..3 (i,f) and 6,7 (o) are pre-scaled by 0.5 for the tanh-based sigmoid.
    // lane L: M = L%16 (row g = 16n+M); VGPR r: K = 16*half + 2r
    v16h a[8];
    #pragma unroll
    for (int n = 0; n < 8; ++n) {
        const float sc = (n == 4 || n == 5) ? 1.0f : 0.5f;
        const float* wr = W_hh + (size_t)(16 * n + brow) * HH;
        #pragma unroll
        for (int r = 0; r < 8; ++r) {
            int k0 = 16 * half + 2 * r;
            a[n][2 * r]     = (_Float16)(wr[k0]     * sc);
            a[n][2 * r + 1] = (_Float16)(wr[k0 + 1] * sc);
        }
    }

    // ------------ packed (bias lo, W_ih hi) f16x2 per gate, D-layout ---------
    // D tile n, VGPR k corresponds to gate g = 16n + k + 8*half  (col = batch)
    unsigned int bw[8][8];
    #pragma unroll
    for (int n = 0; n < 8; ++n) {
        const float sc = (n == 4 || n == 5) ? 1.0f : 0.5f;
        #pragma unroll
        for (int k = 0; k < 8; ++k) {
            int g = 16 * n + k + 8 * half;
            bw[n][k] = pk_f16((b_ih[g] + b_hh[g]) * sc, W_ih[g] * sc);
        }
    }

    // ---------------- recurrent state (h^T / c^T in D-layout) ----------------
    float c0[8], c1[8], ht0[8], ht1[8];   // tile0: hid 0..15 rows, tile1: hid 16..31
    #pragma unroll
    for (int k = 0; k < 8; ++k) { c0[k] = 0.f; c1[k] = 0.f; ht0[k] = 0.f; ht1[k] = 0.f; }

    const float* xrow = x + (size_t)(b0 + brow) * T_STEPS;
    float xcur = xrow[0];

    #pragma unroll 1
    for (int t = 0; t < T_STEPS; ++t) {
        // prefetch next scalar input
        float xnxt = xrow[(t + 1 < T_STEPS) ? (t + 1) : (T_STEPS - 1)];

        // ---- build B operand vb = h^T [32x16] in f16 B-layout --------------
        // half-wave exchange: lanes<16 need partner's ht0 (hid 8..15),
        //                     lanes>=16 need partner's ht1 (hid 16..23)
        float recv[8];
        #pragma unroll
        for (int k = 0; k < 8; ++k) {
            float send = (half == 0) ? ht1[k] : ht0[k];
            recv[k] = __shfl_xor(send, 16, 32);
        }
        union { v16h h; unsigned int u[8]; } vbu;
        #pragma unroll
        for (int r = 0; r < 4; ++r) {
            float l0 = (half == 0) ? ht0[2 * r]     : recv[2 * r];
            float l1 = (half == 0) ? ht0[2 * r + 1] : recv[2 * r + 1];
            vbu.u[r] = pk_f16(l0, l1);                       // K = 2r,2r+1 (+16 on half1)
            float m0 = (half == 0) ? recv[2 * r]     : ht1[2 * r];
            float m1 = (half == 0) ? recv[2 * r + 1] : ht1[2 * r + 1];
            vbu.u[r + 4] = pk_f16(m0, m1);                   // K = 8+2r,9+2r (+16 on half1)
        }
        v16h vb = vbu.h;

        // ---- 8 WMMAs: gates^T tile n = W_hh_tile_n @ h^T + (bias + x*W_ih) -
        v8f dt[8];
        #pragma unroll
        for (int n = 0; n < 8; ++n) {
            v8f cacc;
            #pragma unroll
            for (int k = 0; k < 8; ++k) {
                cacc[k] = gate_init(bw[n][k], xcur);
            }
            dt[n] = __builtin_amdgcn_wmma_f32_16x16x32_f16(
                        false, a[n], false, vb, (short)0, cacc, false, false);
        }

        // ---- activations + state update (tiles 0,1=i 2,3=f 4,5=g 6,7=o) ----
        #pragma unroll
        for (int k = 0; k < 8; ++k) {
            float ig0 = sigm_scaled(dt[0][k]); float ig1 = sigm_scaled(dt[1][k]);
            float fg0 = sigm_scaled(dt[2][k]); float fg1 = sigm_scaled(dt[3][k]);
            float gg0 = tanh_hw(dt[4][k]);     float gg1 = tanh_hw(dt[5][k]);
            float og0 = sigm_scaled(dt[6][k]); float og1 = sigm_scaled(dt[7][k]);
            c0[k]  = fg0 * c0[k] + ig0 * gg0;
            c1[k]  = fg1 * c1[k] + ig1 * gg1;
            ht0[k] = og0 * tanh_hw(c0[k]);
            ht1[k] = og1 * tanh_hw(c1[k]);
        }
        xcur = xnxt;
    }

    // ---------------- head: relu(h @ W3^T + b3) @ W4^T -> sigmoid ------------
    float* hb = hbuf + wave * 16 * HH;
    #pragma unroll
    for (int k = 0; k < 8; ++k) {
        hb[brow * HH + (k + 8 * half)]      = ht0[k];
        hb[brow * HH + 16 + (k + 8 * half)] = ht1[k];
    }
    __syncthreads();

    if (lane < 16) {
        const float* hv = hb + lane * HH;
        float z = b4[0];
        #pragma unroll 4
        for (int j = 0; j < HH; ++j) {
            float acc = b3[j];
            const float* w3r = W3 + j * HH;
            #pragma unroll
            for (int k = 0; k < HH; ++k) acc = fmaf(hv[k], w3r[k], acc);
            acc = fmaxf(acc, 0.0f);
            z = fmaf(acc, W4[j], z);
        }
        out[b0 + lane] = fmaf(tanh_hw(0.5f * z), 0.5f, 0.5f);   // sigmoid(z)
    }
}

extern "C" void kernel_launch(void* const* d_in, const int* in_sizes, int n_in,
                              void* d_out, int out_size, void* d_ws, size_t ws_size,
                              hipStream_t stream) {
    (void)n_in; (void)d_ws; (void)ws_size; (void)out_size;
    const float* x    = (const float*)d_in[0];
    const float* W_ih = (const float*)d_in[1];
    const float* W_hh = (const float*)d_in[2];
    const float* b_ih = (const float*)d_in[3];
    const float* b_hh = (const float*)d_in[4];
    const float* W3   = (const float*)d_in[5];
    const float* b3   = (const float*)d_in[6];
    const float* W4   = (const float*)d_in[7];
    const float* b4   = (const float*)d_in[8];
    float* out = (float*)d_out;

    const int Bsz = in_sizes[0] / T_STEPS;          // 8192
    dim3 grid(Bsz / (16 * 8));                      // 16 batch/wave, 8 waves/block -> 64 blocks
    lstm_wmma_kernel<<<grid, 256, 0, stream>>>(x, W_ih, W_hh, b_ih, b_hh,
                                               W3, b3, W4, b4, out);
}
